// MultiHeadSelfAttention_68891275428516
// MI455X (gfx1250) — compile-verified, matrix operand order checked by reading, not run on hardware
//
#include <hip/hip_runtime.h>
#include <hip/hip_bf16.h>
#include <math.h>

// ---------------------------------------------------------------------------
// MHA + RoPE for B=1, S=4096, D=768, H=12, dh=64 on gfx1250 (wave32, WMMA).
//   0) f32 -> bf16 conversion of x and weights
//   1) QKV projection (v_wmma_f32_16x16x32_bf16) with fused RoPE;
//      Q/K -> [h][s][64], V -> transposed [h][64][s] (b128 stores)
//   2) flash attention: double-buffered global_load_async_to_lds_b128 staging
//      of K/V tiles, WMMA QK^T and PV from LDS, online softmax in f32
//   3) output projection WMMA -> f32
// ---------------------------------------------------------------------------

typedef __bf16 bf16;
typedef __attribute__((ext_vector_type(16))) __bf16 v16bf;
typedef __attribute__((ext_vector_type(8)))  __bf16 v8bf;
typedef __attribute__((ext_vector_type(8)))  float  v8f;

#define S_LEN 4096
#define D_DIM 768
#define NH    12
#define DH    64

static __device__ __forceinline__ int lane_id() { return threadIdx.x & 31; }

// 16x32 bf16 tile, "row" (A-row or B-column) = lane&15, contraction dim
// contiguous in memory (ISA 7.12.2 16-bit A layout; B mirrors it per column).
static __device__ __forceinline__ v16bf load_tile_rowmajor(const bf16* p, int ld) {
  const int l  = lane_id();
  const int r  = l & 15;
  const int kh = (l >> 4) << 3;
  const bf16* q = p + (size_t)r * ld + kh;
  v16bf v;
#pragma unroll
  for (int j = 0; j < 8; ++j) v[j] = q[j];
#pragma unroll
  for (int j = 0; j < 8; ++j) v[8 + j] = q[16 + j];
  return v;
}

// Async copy 16 bytes global -> LDS (ASYNCcnt-tracked, CDNA5 §10/§15.18).
static __device__ __forceinline__ void async_b128(unsigned ldsoff, const void* g) {
  unsigned long long ga = (unsigned long long)g;
  asm volatile("global_load_async_to_lds_b128 %0, %1, off"
               :: "v"(ldsoff), "v"(ga) : "memory");
}
static __device__ __forceinline__ void wait_async0() {
  asm volatile("s_wait_asynccnt 0x0" ::: "memory");
}

// ---------------------------------------------------------------------------
__global__ void cvt_f32_bf16(const float* __restrict__ src, bf16* __restrict__ dst, int n) {
  int i = blockIdx.x * blockDim.x + threadIdx.x;
  if (i < n) dst[i] = (bf16)src[i];
}

// ---------------------------------------------------------------------------
// QKV projection + fused RoPE.  grid = (D/64, S/64, 3), block = 128 (4 waves).
__global__ __launch_bounds__(128) void qkv_rope_kernel(
    const bf16* __restrict__ Xb, const bf16* __restrict__ Wqb,
    const bf16* __restrict__ Wkb, const bf16* __restrict__ Wvb,
    const int* __restrict__ pos,
    bf16* __restrict__ Qb, bf16* __restrict__ Kb, bf16* __restrict__ Vt) {
  const int wave  = threadIdx.x >> 5;
  const int l     = threadIdx.x & 31;
  const int col0  = blockIdx.x * 64;               // output feature tile
  const int row0  = blockIdx.y * 64 + wave * 16;   // token rows for this wave
  const int which = blockIdx.z;                    // 0=Q 1=K 2=V
  const bf16* W   = (which == 0) ? Wqb : (which == 1) ? Wkb : Wvb;

  v8f acc[4] = {};
  for (int k = 0; k < D_DIM; k += 32) {
    __builtin_prefetch(Xb + (size_t)row0 * D_DIM + k + 128, 0, 1);
    v16bf a = load_tile_rowmajor(Xb + (size_t)row0 * D_DIM + k, D_DIM);
#pragma unroll
    for (int t = 0; t < 4; ++t) {
      // einsum 'sd,kd->sk': B column n == W row n, contiguous in d.
      v16bf b = load_tile_rowmajor(W + (size_t)(col0 + t * 16) * D_DIM + k, D_DIM);
      acc[t] = __builtin_amdgcn_wmma_f32_16x16x32_bf16(
          false, a, false, b, (short)0, acc[t], false, false);
    }
  }

  const int coll  = l & 15;
  const int rbase = (l >> 4) * 8;
#pragma unroll
  for (int t = 0; t < 4; ++t) {
    const int n = col0 + t * 16 + coll;   // global feature index
    const int h = n >> 6;                 // head
    const int c = n & 63;                 // index within head
    if (which == 2) {
      // V stored transposed [h][dh][s]: 8 consecutive rows -> one b128 store
      v8bf vv;
#pragma unroll
      for (int r = 0; r < 8; ++r) vv[r] = (bf16)acc[t][r];
      *(v8bf*)&Vt[((size_t)h * DH + c) * S_LEN + row0 + rbase] = vv;
    } else {
      bf16* Out = (which == 0) ? Qb : Kb;
      // theta^(-2i/dh) = exp(-(2i) * ln(1e4)/64); hoisted out of the r-loop
      const float i2   = (float)(c & ~1);
      const float freq = __expf(i2 * -0.14391156f);
#pragma unroll
      for (int r = 0; r < 8; ++r) {
        const int srow = row0 + rbase + r;
        float v = acc[t][r];
        // pair partner (col ^ 1) lives in lane ^ 1 of the C fragment
        float partner = __shfl_xor(v, 1, 32);
        float sn, cs;
        __sincosf((float)pos[srow] * freq, &sn, &cs);
        v = (c & 1) ? fmaf(partner, sn, v * cs)    // r2 = x1*sin + x2*cos
                    : fmaf(v, cs, -partner * sn);  // r1 = x1*cos - x2*sin
        Out[((size_t)h * S_LEN + srow) * DH + c] = (bf16)v;
      }
    }
  }
}

// ---------------------------------------------------------------------------
// Stage one 32-key K tile (32x64) and V tile (64x32, transposed layout) into
// LDS with async b128 copies; 128 threads x 2 chunks each per tile.
static __device__ __forceinline__ void stage_kv(const bf16* __restrict__ Kh,
                                                const bf16* __restrict__ Vh,
                                                bf16* ldsK, bf16* ldsV,
                                                int kbase, int tid) {
  const int c0 = tid * 2;
#pragma unroll
  for (int u = 0; u < 2; ++u) {
    const int c = c0 + u;
    {  // K: row = key (64B*2 per row => 8 chunks), contiguous in d
      const int row = c >> 3, off = (c & 7) * 8;
      async_b128((unsigned)(uintptr_t)(ldsK + row * 64 + off),
                 Kh + (size_t)(kbase + row) * DH + off);
    }
    {  // V (transposed [dh][s]): row = dh (4 chunks per row), contiguous keys
      const int row = c >> 2, off = (c & 3) * 8;
      async_b128((unsigned)(uintptr_t)(ldsV + row * 32 + off),
                 Vh + (size_t)row * S_LEN + kbase + off);
    }
  }
}

// ---------------------------------------------------------------------------
// Flash attention, causal.  grid = (S/64, H), block = 128 (4 waves).
// K/V tiles double-buffered in LDS via global_load_async_to_lds_b128.
__global__ __launch_bounds__(128) void flash_attn_kernel(
    const bf16* __restrict__ Qb, const bf16* __restrict__ Kb,
    const bf16* __restrict__ Vt, bf16* __restrict__ Ob) {
  __shared__ bf16 ldsK[2][32 * 64];   // [buf][key][d]
  __shared__ bf16 ldsV[2][64 * 32];   // [buf][d][key]
  __shared__ bf16 ldsP[4][16 * 32];   // P staging per wave

  const int tid   = threadIdx.x;
  const int wave  = tid >> 5;
  const int l     = tid & 31;
  const int h     = blockIdx.y;
  const int qblk  = blockIdx.x * 64;
  const int q0    = qblk + wave * 16;
  const int coll  = l & 15;
  const int rbase = (l >> 4) * 8;

  const bf16* Qh = Qb + (size_t)h * S_LEN * DH;
  const bf16* Kh = Kb + (size_t)h * S_LEN * DH;
  const bf16* Vh = Vt + (size_t)h * DH * S_LEN;

  // Q tile fragments (A-layout), d-chunks 0..31 and 32..63
  const v16bf aq0 = load_tile_rowmajor(Qh + (size_t)q0 * DH + 0,  DH);
  const v16bf aq1 = load_tile_rowmajor(Qh + (size_t)q0 * DH + 32, DH);

  float m_i[8], l_i[8];
  v8f   o_acc[4] = {};
#pragma unroll
  for (int r = 0; r < 8; ++r) { m_i[r] = -INFINITY; l_i[r] = 0.0f; }

  const float scale = 0.125f;  // 1/sqrt(64)
  bf16* myp = &ldsP[wave][0];

  const int ntiles = (qblk + 64) / 32;   // block-uniform causal extent
  stage_kv(Kh, Vh, &ldsK[0][0], &ldsV[0][0], 0, tid);

  for (int kt = 0; kt < ntiles; ++kt) {
    const int buf   = kt & 1;
    const int kbase = kt * 32;

    wait_async0();        // this wave's chunks of tile kt landed
    __syncthreads();      // everyone's chunks landed; prev buf free to reuse

    if (kt + 1 < ntiles)  // prefetch next tile into the other buffer
      stage_kv(Kh, Vh, &ldsK[buf ^ 1][0], &ldsV[buf ^ 1][0], kbase + 32, tid);

    if (kbase <= q0 + 15) {  // wave-uniform: tile intersects causal window
      // S = Q K^T from LDS K tile
      v8f s0 = {}, s1 = {};
      {
        v16bf b0 = load_tile_rowmajor(&ldsK[buf][0], 64);
        v16bf b1 = load_tile_rowmajor(&ldsK[buf][32], 64);
        s0 = __builtin_amdgcn_wmma_f32_16x16x32_bf16(false, aq0, false, b0, (short)0, s0, false, false);
        s0 = __builtin_amdgcn_wmma_f32_16x16x32_bf16(false, aq1, false, b1, (short)0, s0, false, false);
      }
      {
        v16bf b0 = load_tile_rowmajor(&ldsK[buf][16 * 64], 64);
        v16bf b1 = load_tile_rowmajor(&ldsK[buf][16 * 64 + 32], 64);
        s1 = __builtin_amdgcn_wmma_f32_16x16x32_bf16(false, aq0, false, b0, (short)0, s1, false, false);
        s1 = __builtin_amdgcn_wmma_f32_16x16x32_bf16(false, aq1, false, b1, (short)0, s1, false, false);
      }

      // scale + causal mask + row-max (reduce over 16 lanes of each half)
      float rm[8];
#pragma unroll
      for (int r = 0; r < 8; ++r) {
        const int qrow = q0 + rbase + r;
        float v0 = (kbase + coll      <= qrow) ? s0[r] * scale : -INFINITY;
        float v1 = (kbase + 16 + coll <= qrow) ? s1[r] * scale : -INFINITY;
        s0[r] = v0; s1[r] = v1;
        float mx = fmaxf(v0, v1);
#pragma unroll
        for (int off = 8; off >= 1; off >>= 1)
          mx = fmaxf(mx, __shfl_xor(mx, off, 16));
        rm[r] = mx;
      }

      // online softmax update + stage P into LDS as bf16
#pragma unroll
      for (int r = 0; r < 8; ++r) {
        const float mnew = fmaxf(m_i[r], rm[r]);
        const float corr = __expf(m_i[r] - mnew);   // 0 when m_i == -inf
        const float p0   = __expf(s0[r] - mnew);    // masked -> exp(-inf) = 0
        const float p1   = __expf(s1[r] - mnew);
        float rs = p0 + p1;
#pragma unroll
        for (int off = 8; off >= 1; off >>= 1)
          rs += __shfl_xor(rs, off, 16);
        l_i[r] = l_i[r] * corr + rs;
        m_i[r] = mnew;
#pragma unroll
        for (int t = 0; t < 4; ++t) o_acc[t][r] *= corr;
        myp[(rbase + r) * 32 + coll]      = (bf16)p0;
        myp[(rbase + r) * 32 + 16 + coll] = (bf16)p1;
      }

      // O += P V  (P re-loaded in A-layout; V tile is [dh][key] in LDS)
      v16bf pa = load_tile_rowmajor(myp, 32);
#pragma unroll
      for (int t = 0; t < 4; ++t) {
        v16bf bv = load_tile_rowmajor(&ldsV[buf][t * 16 * 32], 32);
        o_acc[t] = __builtin_amdgcn_wmma_f32_16x16x32_bf16(
            false, pa, false, bv, (short)0, o_acc[t], false, false);
      }
    }
  }

  // normalize and store O as bf16 in [s][D] layout for the output projection
#pragma unroll
  for (int r = 0; r < 8; ++r) {
    const int srow = q0 + rbase + r;
    const float inv = 1.0f / l_i[r];
#pragma unroll
    for (int t = 0; t < 4; ++t)
      Ob[(size_t)srow * D_DIM + h * DH + t * 16 + coll] = (bf16)(o_acc[t][r] * inv);
  }
}

// ---------------------------------------------------------------------------
// Output projection: out[s,m] = sum_v O[s,v] * Wo[m,v].  grid = (D/64, S/64).
__global__ __launch_bounds__(128) void out_proj_kernel(
    const bf16* __restrict__ Ob, const bf16* __restrict__ Wob,
    float* __restrict__ out) {
  const int wave = threadIdx.x >> 5;
  const int l    = threadIdx.x & 31;
  const int col0 = blockIdx.x * 64;
  const int row0 = blockIdx.y * 64 + wave * 16;

  v8f acc[4] = {};
  for (int k = 0; k < D_DIM; k += 32) {
    __builtin_prefetch(Ob + (size_t)row0 * D_DIM + k + 128, 0, 1);
    v16bf a = load_tile_rowmajor(Ob + (size_t)row0 * D_DIM + k, D_DIM);
#pragma unroll
    for (int t = 0; t < 4; ++t) {
      v16bf b = load_tile_rowmajor(Wob + (size_t)(col0 + t * 16) * D_DIM + k, D_DIM);
      acc[t] = __builtin_amdgcn_wmma_f32_16x16x32_bf16(
          false, a, false, b, (short)0, acc[t], false, false);
    }
  }

  const int coll  = l & 15;
  const int rbase = (l >> 4) * 8;
#pragma unroll
  for (int t = 0; t < 4; ++t)
#pragma unroll
    for (int r = 0; r < 8; ++r)
      out[(size_t)(row0 + rbase + r) * D_DIM + col0 + t * 16 + coll] = acc[t][r];
}

// ---------------------------------------------------------------------------
extern "C" void kernel_launch(void* const* d_in, const int* in_sizes, int n_in,
                              void* d_out, int out_size, void* d_ws, size_t ws_size,
                              hipStream_t stream) {
  const float* x   = (const float*)d_in[0];
  const int*   pos = (const int*)  d_in[1];
  const float* Wq  = (const float*)d_in[2];
  const float* Wk  = (const float*)d_in[3];
  const float* Wv  = (const float*)d_in[4];
  const float* Wo  = (const float*)d_in[5];
  float* out = (float*)d_out;

  const size_t nXD = (size_t)S_LEN * D_DIM;   // 3,145,728
  const size_t nW  = (size_t)D_DIM * D_DIM;   //   589,824

  char* ws = (char*)d_ws;
  size_t off = 0;
  auto carve = [&](size_t bytes) {
    char* p = ws + off;
    off += (bytes + 255) & ~(size_t)255;
    return p;
  };
  bf16* Xb  = (bf16*)carve(nXD * sizeof(bf16));
  bf16* Wqb = (bf16*)carve(nW * sizeof(bf16));
  bf16* Wkb = (bf16*)carve(nW * sizeof(bf16));
  bf16* Wvb = (bf16*)carve(nW * sizeof(bf16));
  bf16* Wob = (bf16*)carve(nW * sizeof(bf16));
  bf16* Qb  = (bf16*)carve(nXD * sizeof(bf16));  // [h][s][64]
  bf16* Kb  = (bf16*)carve(nXD * sizeof(bf16));  // [h][s][64]
  bf16* Vt  = (bf16*)carve(nXD * sizeof(bf16));  // [h][64][s] (transposed)
  bf16* Ob  = (bf16*)carve(nXD * sizeof(bf16));  // [s][768]

  // 0) f32 -> bf16 conversions
  cvt_f32_bf16<<<(int)((nXD + 255) / 256), 256, 0, stream>>>(x,  Xb,  (int)nXD);
  cvt_f32_bf16<<<(int)((nW  + 255) / 256), 256, 0, stream>>>(Wq, Wqb, (int)nW);
  cvt_f32_bf16<<<(int)((nW  + 255) / 256), 256, 0, stream>>>(Wk, Wkb, (int)nW);
  cvt_f32_bf16<<<(int)((nW  + 255) / 256), 256, 0, stream>>>(Wv, Wvb, (int)nW);
  cvt_f32_bf16<<<(int)((nW  + 255) / 256), 256, 0, stream>>>(Wo, Wob, (int)nW);

  // 1) QKV projection + RoPE (V written transposed)
  qkv_rope_kernel<<<dim3(D_DIM / 64, S_LEN / 64, 3), 128, 0, stream>>>(
      Xb, Wqb, Wkb, Wvb, pos, Qb, Kb, Vt);

  // 2) causal flash attention with async LDS staging
  flash_attn_kernel<<<dim3(S_LEN / 64, NH), 128, 0, stream>>>(Qb, Kb, Vt, Ob);

  // 3) output projection -> f32
  out_proj_kernel<<<dim3(D_DIM / 64, S_LEN / 64), 128, 0, stream>>>(Ob, Wob, out);
}
